// AttentionConv2D_82377472737620
// MI455X (gfx1250) — matrix operand order repaired and checked by reading, not compiled, not run on hardware
//
#include <hip/hip_runtime.h>
#include <math.h>

typedef __attribute__((ext_vector_type(16))) _Float16 v16h;
typedef __attribute__((ext_vector_type(8)))  _Float16 v8h;
typedef __attribute__((ext_vector_type(8)))  float    v8f;

#define DEVI __device__ __forceinline__

// Problem constants
constexpr int Bn  = 4;
constexpr int C   = 64;
constexpr int H   = 128;
constexpr int Wd  = 128;
constexpr int Ad  = 64;    // attention size
constexpr int HID = 128;   // 2*A
constexpr int OUT = 64;
constexpr int NPIX = Bn * H * Wd;        // 65536
constexpr float SCALE = 0.125f;          // 64^-0.5

// LDS strides (halves). All fragment accesses land on 16B boundaries:
// 72 halves = 144B = 16*9 ; 136 halves = 272B = 16*17 ; 40 halves = 80B = 16*5
constexpr int XS  = 72;    // Xn row stride
constexpr int W1S = 72;    // W1t row stride (row = output col n, 64 K entries)
constexpr int W2S = 136;   // W2t row stride (row = output col n, 128 K entries)
constexpr int HS  = 40;    // hidden-chunk row stride (32 K entries)

// ---- WMMA fragment loaders (ISA 7.12.2 layouts, 16-bit, 16x16x32) ----
// A (16xK=32): lanes 0-15 hold row=lane, halves {K0..K0+7, K0+16..K0+23};
//              lanes 16-31 hold row=lane-16, halves {K0+8..K0+15, K0+24..K0+31}.
DEVI v16h frag_A(const _Float16* rowbase, int k0, int lane) {
  int kb = k0 + ((lane & 16) ? 8 : 0);
  v8h lo = *(const v8h*)(rowbase + kb);
  v8h hi = *(const v8h*)(rowbase + kb + 16);
  union { v16h v; v8h h[2]; } u; u.h[0] = lo; u.h[1] = hi;
  return u.v;
}
// B (K=32 x 16), stored transposed in LDS as [n][k]: lanes 0-15 hold col=lane,
// halves {K0..K0+7, K0+16..K0+23}; lanes 16-31 hold col=lane-16, {K0+8..K0+15, K0+24..K0+31}.
DEVI v16h frag_B(const _Float16* rowbase, int k0, int lane) {
  int kb = k0 + ((lane & 16) ? 8 : 0);
  v8h lo = *(const v8h*)(rowbase + kb);
  v8h hi = *(const v8h*)(rowbase + kb + 16);
  union { v16h v; v8h h[2]; } u; u.h[0] = lo; u.h[1] = hi;
  return u.v;
}

DEVI v8f wmma_f16(v16h a, v16h b, v8f c) {
  return __builtin_amdgcn_wmma_f32_16x16x32_f16(false, a, false, b, (short)0, c,
                                                false, false);
}

// ============================================================
// Kernel 1: positional key bias  posk[p][a], p in 0..8
// ============================================================
__global__ void posk_kernel(const float* __restrict__ Wpos,
                            const float* __restrict__ bpos,
                            const float* __restrict__ Wposk,
                            const float* __restrict__ bposk,
                            float* __restrict__ poskw) {
  __shared__ float pemb[9 * 64];
  const int a = threadIdx.x;          // 0..63
  const float kLn = 0.14391156f;      // ln(10000)/64
  if (a < 64) {
    for (int p = 0; p < 9; ++p) {
      float acc = bpos[a];
#pragma unroll 8
      for (int c = 0; c < 64; ++c) {
        float phase = (float)p * __expf(-(float)(c & ~1) * kLn);
        float pe = (c & 1) ? __cosf(phase) : __sinf(phase);
        acc += pe * Wpos[c * 64 + a];
      }
      pemb[p * 64 + a] = fmaxf(acc, 0.f);
    }
  }
  __syncthreads();
  if (a < 64) {
    for (int p = 0; p < 9; ++p) {
      float acc = bposk[a];
#pragma unroll 8
      for (int j = 0; j < 64; ++j) acc += pemb[p * 64 + j] * Wposk[j * 64 + a];
      poskw[p * 64 + a] = acc;
    }
  }
}

// ============================================================
// Kernel 2: LayerNorm + fused Q/K/V 2-layer MLPs via WMMA
// grid = 512 blocks (one (b,h) row, 128 pixels), 256 threads = 8 waves
// ============================================================
__global__ __launch_bounds__(256) void lnqkv_kernel(
    const float* __restrict__ x, const float* __restrict__ ln_g,
    const float* __restrict__ ln_b,
    const float* __restrict__ Wq, const float* __restrict__ bq,
    const float* __restrict__ Wq2, const float* __restrict__ bq2,
    const float* __restrict__ Wk, const float* __restrict__ bk,
    const float* __restrict__ Wk2, const float* __restrict__ bk2,
    const float* __restrict__ Wv, const float* __restrict__ bv,
    const float* __restrict__ Wv2, const float* __restrict__ bv2,
    float* __restrict__ Qw, float* __restrict__ Kw, float* __restrict__ Vw) {
  __shared__ __align__(16) _Float16 Xn[128 * XS];      // 18432 B
  __shared__ __align__(16) _Float16 W1t[128 * W1S];    // 18432 B
  __shared__ __align__(16) _Float16 W2t[64 * W2S];     // 17408 B
  __shared__ __align__(16) _Float16 hidc[8 * 16 * HS]; // 10240 B
  __shared__ float b1s[HID];
  __shared__ float b2s[OUT];

  const int tid = threadIdx.x;
  const int bid = blockIdx.x;           // 0..511
  const int bb = bid >> 7;              // batch
  const int hh = bid & 127;             // row
  const int pixBase = bid * Wd;         // flat pixel index of w=0

  // ---- LayerNorm: threads 0..127 handle one pixel each ----
  if (tid < Wd) {
    const size_t base = (size_t)bb * C * H * Wd + (size_t)hh * Wd + tid;
    float xv[64];
    float s = 0.f;
#pragma unroll
    for (int c = 0; c < C; ++c) { xv[c] = x[base + (size_t)c * (H * Wd)]; s += xv[c]; }
    float mu = s * (1.f / 64.f);
    float ss = 0.f;
#pragma unroll
    for (int c = 0; c < C; ++c) { float d = xv[c] - mu; ss += d * d; }
    float inv = __frsqrt_rn(ss * (1.f / 64.f) + 1e-5f);
#pragma unroll
    for (int c = 0; c < C; ++c)
      Xn[tid * XS + c] = (_Float16)((xv[c] - mu) * inv * ln_g[c] + ln_b[c]);
  }

  const int wave = tid >> 5;
  const int lane = tid & 31;
  const int m0 = wave * 16;
  const int col = lane & 15;
  const int rsel = (lane & 16) ? 8 : 0;
  _Float16* hw = &hidc[wave * 16 * HS];

  for (int j = 0; j < 3; ++j) {
    const float* W1g = (j == 0) ? Wq : (j == 1) ? Wk : Wv;
    const float* b1g = (j == 0) ? bq : (j == 1) ? bk : bv;
    const float* W2g = (j == 0) ? Wq2 : (j == 1) ? Wk2 : Wv2;
    const float* b2g = (j == 0) ? bq2 : (j == 1) ? bk2 : bv2;
    float* outg = (j == 0) ? Qw : (j == 1) ? Kw : Vw;

    __syncthreads();  // previous iter's LDS reads done (and Xn ready at j==0)
    // stage weights transposed as f16: W1t[n][c], W2t[n][k]
    for (int i = tid; i < C * HID; i += 256) {
      int c = i >> 7, n = i & 127;
      W1t[n * W1S + c] = (_Float16)W1g[i];
    }
    for (int i = tid; i < HID * OUT; i += 256) {
      int k = i >> 6, n = i & 63;
      W2t[n * W2S + k] = (_Float16)W2g[i];
    }
    if (tid < HID) b1s[tid] = b1g[tid];
    if (tid < OUT) b2s[tid] = b2g[tid];
    __syncthreads();

    // ---- per-wave 16-row M-tile ----
    v8f acc2[4];
#pragma unroll
    for (int t = 0; t < 4; ++t) {
      float bv2v = b2s[t * 16 + col];
#pragma unroll
      for (int r = 0; r < 8; ++r) acc2[t][r] = bv2v;
    }
    const _Float16* arow = &Xn[(m0 + col) * XS];

#pragma unroll
    for (int kc = 0; kc < 4; ++kc) {
      const int n0 = kc * 32;
      v8f accH[2];
#pragma unroll
      for (int t = 0; t < 2; ++t) {
        float bv1v = b1s[n0 + t * 16 + col];
#pragma unroll
        for (int r = 0; r < 8; ++r) accH[t][r] = bv1v;
      }
#pragma unroll
      for (int k0 = 0; k0 < 64; k0 += 32) {
        v16h a = frag_A(arow, k0, lane);
#pragma unroll
        for (int t = 0; t < 2; ++t) {
          v16h bf = frag_B(&W1t[(n0 + 16 * t + col) * W1S], k0, lane);
          accH[t] = wmma_f16(a, bf, accH[t]);
        }
      }
      // relu -> f16 -> per-wave LDS chunk (re-fragment for GEMM2)
#pragma unroll
      for (int t = 0; t < 2; ++t)
#pragma unroll
        for (int r = 0; r < 8; ++r)
          hw[(rsel + r) * HS + t * 16 + col] = (_Float16)fmaxf(accH[t][r], 0.f);

      v16h a2 = frag_A(&hw[col * HS], 0, lane);
#pragma unroll
      for (int t = 0; t < 4; ++t) {
        v16h bf = frag_B(&W2t[(t * 16 + col) * W2S], n0, lane);
        acc2[t] = wmma_f16(a2, bf, acc2[t]);
      }
    }
    // store 16x64 output tile, row-major [pixel][64]
#pragma unroll
    for (int t = 0; t < 4; ++t)
#pragma unroll
      for (int r = 0; r < 8; ++r) {
        int pix = pixBase + m0 + rsel + r;
        outg[(size_t)pix * Ad + t * 16 + col] = acc2[t][r];
      }
  }
}

// ============================================================
// Kernel 3: 3x3 neighborhood attention (softmax over 9)
// one thread per pixel; 256 blocks x 256 threads
// ============================================================
__global__ __launch_bounds__(256) void attn_kernel(
    const float* __restrict__ Qw, const float* __restrict__ Kw,
    const float* __restrict__ Vw, const float* __restrict__ poskw,
    float* __restrict__ out) {
  const int pix = blockIdx.x * 256 + threadIdx.x;
  const int bb = pix >> 14;
  const int rem = pix & 16383;
  const int hh = rem >> 7;
  const int ww = rem & 127;

  float4 q[16];
  const float4* qp = (const float4*)(Qw + (size_t)pix * Ad);
#pragma unroll
  for (int i = 0; i < 16; ++i) q[i] = qp[i];

  float att[9];
  int   npx[9];
  bool  inb[9];
#pragma unroll
  for (int p = 0; p < 9; ++p) {
    const int dh = p / 3 - 1, dw = p % 3 - 1;
    const int nh = hh + dh, nw = ww + dw;
    inb[p] = (nh >= 0) & (nh < H) & (nw >= 0) & (nw < Wd);
    npx[p] = pix + dh * Wd + dw;
    // positional part of the logit is present even for zero-padded neighbors
    float s = 0.f;
    const float4* pk = (const float4*)(poskw + p * Ad);
#pragma unroll
    for (int i = 0; i < 16; ++i) {
      float4 k4 = pk[i];
      s += q[i].x * k4.x + q[i].y * k4.y + q[i].z * k4.z + q[i].w * k4.w;
    }
    if (inb[p]) {
      const float4* kp = (const float4*)(Kw + (size_t)npx[p] * Ad);
#pragma unroll
      for (int i = 0; i < 16; ++i) {
        float4 k4 = kp[i];
        s += q[i].x * k4.x + q[i].y * k4.y + q[i].z * k4.z + q[i].w * k4.w;
      }
    }
    att[p] = s * SCALE;
  }
  float m = att[0];
#pragma unroll
  for (int p = 1; p < 9; ++p) m = fmaxf(m, att[p]);
  float den = 0.f;
#pragma unroll
  for (int p = 0; p < 9; ++p) { att[p] = __expf(att[p] - m); den += att[p]; }
  float rden = __frcp_rn(den);
#pragma unroll
  for (int p = 0; p < 9; ++p) att[p] *= rden;

  const size_t obase = (size_t)bb * OUT * H * Wd + (size_t)hh * Wd + ww;
#pragma unroll
  for (int i = 0; i < 16; ++i) {
    float4 acc = make_float4(0.f, 0.f, 0.f, 0.f);
#pragma unroll
    for (int p = 0; p < 9; ++p) {
      if (inb[p]) {
        float4 v4 = *(const float4*)(Vw + (size_t)npx[p] * Ad + i * 4);
        acc.x += att[p] * v4.x; acc.y += att[p] * v4.y;
        acc.z += att[p] * v4.z; acc.w += att[p] * v4.w;
      }
    }
    out[obase + (size_t)(4 * i + 0) * (H * Wd)] = acc.x;
    out[obase + (size_t)(4 * i + 1) * (H * Wd)] = acc.y;
    out[obase + (size_t)(4 * i + 2) * (H * Wd)] = acc.z;
    out[obase + (size_t)(4 * i + 3) * (H * Wd)] = acc.w;
  }
}

// ============================================================
extern "C" void kernel_launch(void* const* d_in, const int* in_sizes, int n_in,
                              void* d_out, int out_size, void* d_ws, size_t ws_size,
                              hipStream_t stream) {
  (void)in_sizes; (void)n_in; (void)out_size; (void)ws_size;
  const float* x    = (const float*)d_in[0];
  const float* ln_g = (const float*)d_in[1];
  const float* ln_b = (const float*)d_in[2];
  const float* Wq   = (const float*)d_in[3];
  const float* bq   = (const float*)d_in[4];
  const float* Wq2  = (const float*)d_in[5];
  const float* bq2  = (const float*)d_in[6];
  const float* Wk   = (const float*)d_in[7];
  const float* bk   = (const float*)d_in[8];
  const float* Wk2  = (const float*)d_in[9];
  const float* bk2  = (const float*)d_in[10];
  const float* Wv   = (const float*)d_in[11];
  const float* bv   = (const float*)d_in[12];
  const float* Wv2  = (const float*)d_in[13];
  const float* bv2  = (const float*)d_in[14];
  const float* Wpos = (const float*)d_in[15];
  const float* bpos = (const float*)d_in[16];
  const float* Wposk= (const float*)d_in[17];
  const float* bposk= (const float*)d_in[18];
  float* out = (float*)d_out;

  char* ws = (char*)d_ws;
  float* poskw = (float*)ws;                                   // 9*64 f32
  float* Qw = (float*)(ws + 4096);                             // 16 MB
  float* Kw = (float*)(ws + 4096 + (size_t)NPIX * Ad * 4);     // 16 MB
  float* Vw = (float*)(ws + 4096 + (size_t)NPIX * Ad * 8);     // 16 MB

  posk_kernel<<<1, 64, 0, stream>>>(Wpos, bpos, Wposk, bposk, poskw);
  lnqkv_kernel<<<512, 256, 0, stream>>>(x, ln_g, ln_b,
                                        Wq, bq, Wq2, bq2,
                                        Wk, bk, Wk2, bk2,
                                        Wv, bv, Wv2, bv2,
                                        Qw, Kw, Vw);
  attn_kernel<<<256, 256, 0, stream>>>(Qw, Kw, Vw, poskw, out);
}
